// Attention_80779744903306
// MI455X (gfx1250) — compile-verified, matrix operand order checked by reading, not run on hardware
//
#include <hip/hip_runtime.h>
#include <hip/hip_bf16.h>

typedef __attribute__((ext_vector_type(16))) _Float16 v16h;
typedef __attribute__((ext_vector_type(8)))  _Float16 v8h;
typedef __attribute__((ext_vector_type(8)))  float    v8f;
typedef int v4i __attribute__((vector_size(16)));   // pointee type of the async builtin

#define HEADS 12
#define NSEQ  1024
#define DMODEL 768
#define DHEAD 64
#define BH    96   // 8 * 12

// Padded LDS leading dims (f16 elements). All chosen so row stride is a
// multiple of 16B (b128-aligned) and NOT ≡ 0 mod 256B (bank-conflict-free
// for the 16-lane strided ds_read_b128 fragment pattern).
#define LDA_SA  40    // 32-deep GEMM tiles:   80B stride  (20 banks)
#define LDA_SK  72    // K tile (d=64 deep):  144B stride  (36 banks)
#define LDA_SV  136   // V/P tiles (128 deep):272B stride  ( 4 banks mod 64)

__device__ __forceinline__ v8f wmma16(v16h a, v16h b, v8f c) {
    return __builtin_amdgcn_wmma_f32_16x16x32_f16(
        false, a, false, b, (short)0, c, false, false);
}

// ---- CDNA5 async global->LDS copy (ASYNCcnt) with portable fallback ----
__device__ __forceinline__ void async_copy16(const _Float16* g, _Float16* l) {
#if __has_builtin(__builtin_amdgcn_global_load_async_to_lds_b128)
    __builtin_amdgcn_global_load_async_to_lds_b128(
        (__attribute__((address_space(1))) v4i*)(void*)g,
        (__attribute__((address_space(3))) v4i*)(void*)l, 0, 0);
#else
    *(v8h*)l = *(const v8h*)g;
#endif
}

__device__ __forceinline__ void async_wait_and_barrier() {
#if __has_builtin(__builtin_amdgcn_s_wait_asynccnt)
    __builtin_amdgcn_s_wait_asynccnt(0);
#endif
    __syncthreads();
}

// Load one 16x32 WMMA fragment (A rows / B-from-transposed columns) that is
// K-contiguous with leading dimension `ld`.  ISA 7.12.2 16-bit layout:
// lane<16 holds K={0..7,16..23}, lane>=16 holds K={8..15,24..31}.
__device__ __forceinline__ v16h load_frag(const _Float16* p, int ld, int k0) {
    const int lane = threadIdx.x & 31;
    const int half = lane >> 4;
    const int idx  = lane & 15;
    const _Float16* base = p + (size_t)idx * ld + k0 + half * 8;
    v8h lo = *(const v8h*)(base);
    v8h hi = *(const v8h*)(base + 16);
    v16h r;
#pragma unroll
    for (int e = 0; e < 8; ++e) { r[e] = lo[e]; r[e + 8] = hi[e]; }
    return r;
}

// ---------------- precision-conversion helpers ----------------

__global__ __launch_bounds__(256) void f32_to_f16_kernel(const float* __restrict__ in,
                                                         _Float16* __restrict__ out, int n) {
    int i = blockIdx.x * blockDim.x + threadIdx.x;
    if (i < n) out[i] = (_Float16)in[i];
}

// in: f32 row-major [K, N] -> out: f16 [N, K]
__global__ __launch_bounds__(256) void transpose_f32_to_f16_kernel(const float* __restrict__ in,
                                                                   _Float16* __restrict__ out,
                                                                   int K, int N) {
    int i = blockIdx.x * blockDim.x + threadIdx.x;
    if (i < K * N) {
        int k = i / N, n = i % N;
        out[(size_t)n * K + k] = (_Float16)in[i];
    }
}

// ---------------- shared LDS-tiled 128x128 GEMM core ----------------
// A: rows row0..row0+127, row-major, ld=K.  BT: cols col0..col0+127 stored
// transposed (K-contiguous), ld=K.  8 waves in 4x2 grid of 32x64 tiles.
__device__ __forceinline__ void gemm_tile_128x128(const _Float16* __restrict__ A,
                                                  const _Float16* __restrict__ BT,
                                                  int K, _Float16* sA, _Float16* sB,
                                                  v8f acc[2][4]) {
    const int tid  = threadIdx.x;
    const int wave = tid >> 5;
    const int wm = wave & 3, wn = wave >> 2;

    for (int k0 = 0; k0 < K; k0 += 32) {
        __syncthreads();                       // previous iter's LDS reads done
        // Stage 128x32 tiles: 512 16B-chunks each, 2 per thread per matrix.
#pragma unroll
        for (int i = 0; i < 2; ++i) {
            int c = tid + i * 256;
            int row = c >> 2, ko = (c & 3) * 8;
            async_copy16(A  + (size_t)row * K + k0 + ko, sA + row * LDA_SA + ko);
        }
#pragma unroll
        for (int i = 0; i < 2; ++i) {
            int c = tid + i * 256;
            int row = c >> 2, ko = (c & 3) * 8;
            async_copy16(BT + (size_t)row * K + k0 + ko, sB + row * LDA_SA + ko);
        }
        async_wait_and_barrier();

        v16h af[2], bf[4];
#pragma unroll
        for (int mi = 0; mi < 2; ++mi)
            af[mi] = load_frag(sA + (wm * 32 + mi * 16) * LDA_SA, LDA_SA, 0);
#pragma unroll
        for (int ni = 0; ni < 4; ++ni)
            bf[ni] = load_frag(sB + (wn * 64 + ni * 16) * LDA_SA, LDA_SA, 0);
#pragma unroll
        for (int mi = 0; mi < 2; ++mi)
#pragma unroll
            for (int ni = 0; ni < 4; ++ni)
                acc[mi][ni] = wmma16(af[mi], bf[ni], acc[mi][ni]);
    }
}

// ---------------- QKV projection GEMM ----------------
// M=8192 N=2304 K=768; grid = 64 * 18, block = 256.
__global__ __launch_bounds__(256) void qkv_gemm_kernel(const _Float16* __restrict__ x,
                                                       const _Float16* __restrict__ wT,
                                                       _Float16* __restrict__ q16,
                                                       _Float16* __restrict__ k16,
                                                       _Float16* __restrict__ vT16) {
    __shared__ _Float16 sA[128 * LDA_SA];
    __shared__ _Float16 sB[128 * LDA_SA];
    const int bm = blockIdx.x % 64;
    const int bn = blockIdx.x / 64;
    const int row0 = bm * 128;
    const int col0 = bn * 128;
    const int wave = threadIdx.x >> 5;
    const int wm = wave & 3, wn = wave >> 2;
    const int lane = threadIdx.x & 31;
    const int half = lane >> 4;
    const int idx  = lane & 15;

    v8f acc[2][4] = {};
    gemm_tile_128x128(x + (size_t)row0 * DMODEL, wT + (size_t)col0 * DMODEL,
                      DMODEL, sA, sB, acc);

    // Epilogue: q/k/v split and head indices are uniform per 16-wide fragment
    // (768 % 16 == 0, 64 % 16 == 0) -> scalar branches, no lane divergence.
    const float SCALE = 0.125f; // 64^-0.5
#pragma unroll
    for (int ni = 0; ni < 4; ++ni) {
        int colbase = col0 + wn * 64 + ni * 16;        // wave-uniform
        int which = colbase / DMODEL;
        int rb    = colbase % DMODEL;
        int hh    = rb >> 6;
        int d     = (rb & 63) + idx;
#pragma unroll
        for (int mi = 0; mi < 2; ++mi)
#pragma unroll
            for (int r = 0; r < 8; ++r) {
                int row = row0 + wm * 32 + mi * 16 + half * 8 + r;  // b*1024+n
                int bb = row >> 10, nn = row & 1023;
                size_t bhx = (size_t)bb * HEADS + hh;
                float v = acc[mi][ni][r];
                if (which == 0)
                    q16[(bhx * NSEQ + nn) * DHEAD + d] = (_Float16)(v * SCALE);
                else if (which == 1)
                    k16[(bhx * NSEQ + nn) * DHEAD + d] = (_Float16)v;
                else
                    vT16[(bhx * DHEAD + d) * NSEQ + nn] = (_Float16)v;
            }
    }
}

// ---------------- flash-style attention ----------------
// grid = 96 (b,h) * 8 (query tiles of 128), block = 256 = 8 waves; K/V tiles
// staged once per block through LDS (async), P re-layout via padded LDS slab.
__global__ __launch_bounds__(256) void attn_kernel(const _Float16* __restrict__ q16,
                                                   const _Float16* __restrict__ k16,
                                                   const _Float16* __restrict__ vT16,
                                                   _Float16* __restrict__ a16) {
    __shared__ _Float16 sK[128 * LDA_SK];          // [key][d]     18.0 KB
    __shared__ _Float16 sV[64 * LDA_SV];           // [d][key]     17.0 KB
    __shared__ _Float16 plds[8][16][LDA_SV];       // wave-private 34.0 KB

    const int bh = blockIdx.x >> 3;
    const int qt = blockIdx.x & 7;
    const int b  = bh / HEADS;
    const int h  = bh % HEADS;
    const int tid  = threadIdx.x;
    const int wave = tid >> 5;
    const int lane = tid & 31;
    const int half = lane >> 4;
    const int idx  = lane & 15;
    const int q0 = qt * 128 + wave * 16;

    // Q fragments are loop-invariant: load once from global.
    const _Float16* qbase = q16 + ((size_t)bh * NSEQ + q0) * DHEAD;
    v16h aq[2];
#pragma unroll
    for (int kk = 0; kk < 2; ++kk) aq[kk] = load_frag(qbase, DHEAD, kk * 32);

    v8f acc_o[4] = {};
    float m_run[8], l_run[8];
#pragma unroll
    for (int r = 0; r < 8; ++r) { m_run[r] = -1e30f; l_run[r] = 0.f; }

    for (int jt = 0; jt < NSEQ; jt += 128) {
        __syncthreads();   // previous iteration's sK/sV reads complete
        // K tile: 128 rows x 64 f16 = 1024 chunks; 4 per thread.
#pragma unroll
        for (int i = 0; i < 4; ++i) {
            int c = tid + i * 256;
            int row = c >> 3, ko = (c & 7) * 8;
            async_copy16(k16 + ((size_t)bh * NSEQ + jt + row) * DHEAD + ko,
                         sK + row * LDA_SK + ko);
        }
        // V tile (transposed): 64 rows x 128 f16 = 1024 chunks.
#pragma unroll
        for (int i = 0; i < 4; ++i) {
            int c = tid + i * 256;
            int row = c >> 4, ko = (c & 15) * 8;
            async_copy16(vT16 + ((size_t)bh * DHEAD + row) * NSEQ + jt + ko,
                         sV + row * LDA_SV + ko);
        }
        async_wait_and_barrier();

        // S(16x128) = Q * K^T from LDS.
        v8f s[8] = {};
#pragma unroll
        for (int kk = 0; kk < 2; ++kk)
#pragma unroll
            for (int nt = 0; nt < 8; ++nt) {
                v16h bk = load_frag(sK + (nt * 16) * LDA_SK, LDA_SK, kk * 32);
                s[nt] = wmma16(aq[kk], bk, s[nt]);
            }

        // Online softmax per row (row per VGPR, columns per lane-of-half).
#pragma unroll
        for (int r = 0; r < 8; ++r) {
            float mx = -1e30f;
#pragma unroll
            for (int nt = 0; nt < 8; ++nt) mx = fmaxf(mx, s[nt][r]);
#pragma unroll
            for (int off = 8; off >= 1; off >>= 1) mx = fmaxf(mx, __shfl_xor(mx, off, 16));
            float mnew = fmaxf(m_run[r], mx);
            float corr = __expf(m_run[r] - mnew);
            float rsum = 0.f;
#pragma unroll
            for (int nt = 0; nt < 8; ++nt) {
                float pe = __expf(s[nt][r] - mnew);
                plds[wave][half * 8 + r][nt * 16 + idx] = (_Float16)pe;
                rsum += pe;
            }
#pragma unroll
            for (int off = 8; off >= 1; off >>= 1) rsum += __shfl_xor(rsum, off, 16);
            l_run[r] = l_run[r] * corr + rsum;
            m_run[r] = mnew;
#pragma unroll
            for (int dn = 0; dn < 4; ++dn) acc_o[dn][r] *= corr;
        }

        // O(16x64) += P(16x128) * V(128x64), both operands from LDS.
#pragma unroll
        for (int kk2 = 0; kk2 < 4; ++kk2) {
            v16h ap = load_frag(&plds[wave][0][0], LDA_SV, kk2 * 32);
#pragma unroll
            for (int dn = 0; dn < 4; ++dn) {
                v16h bv = load_frag(sV + (dn * 16) * LDA_SV, LDA_SV, kk2 * 32);
                acc_o[dn] = wmma16(ap, bv, acc_o[dn]);
            }
        }
    }

    // Normalize, write merged-head layout [b*n, h*64+d] f16 for out-proj.
#pragma unroll
    for (int dn = 0; dn < 4; ++dn)
#pragma unroll
        for (int r = 0; r < 8; ++r) {
            int qrow = q0 + half * 8 + r;
            float o = acc_o[dn][r] / l_run[r];
            a16[((size_t)b * NSEQ + qrow) * DMODEL + h * DHEAD + dn * 16 + idx] = (_Float16)o;
        }
}

// ---------------- output projection GEMM (+bias, f32 out) ----------------
// M=8192 N=768 K=768; grid = 64 * 6, block = 256.
__global__ __launch_bounds__(256) void out_gemm_kernel(const _Float16* __restrict__ a,
                                                       const _Float16* __restrict__ wT,
                                                       const float* __restrict__ bias,
                                                       float* __restrict__ out) {
    __shared__ _Float16 sA[128 * LDA_SA];
    __shared__ _Float16 sB[128 * LDA_SA];
    const int bm = blockIdx.x % 64;
    const int bn = blockIdx.x / 64;
    const int row0 = bm * 128;
    const int col0 = bn * 128;
    const int wave = threadIdx.x >> 5;
    const int wm = wave & 3, wn = wave >> 2;
    const int lane = threadIdx.x & 31;
    const int half = lane >> 4;
    const int idx  = lane & 15;

    v8f acc[2][4] = {};
    gemm_tile_128x128(a + (size_t)row0 * DMODEL, wT + (size_t)col0 * DMODEL,
                      DMODEL, sA, sB, acc);

#pragma unroll
    for (int mi = 0; mi < 2; ++mi)
#pragma unroll
        for (int ni = 0; ni < 4; ++ni) {
            int col = col0 + wn * 64 + ni * 16 + idx;
            float bv = bias[col];
#pragma unroll
            for (int r = 0; r < 8; ++r) {
                int row = row0 + wm * 32 + mi * 16 + half * 8 + r;
                out[(size_t)row * DMODEL + col] = acc[mi][ni][r] + bv;
            }
        }
}

// ---------------- launch ----------------

extern "C" void kernel_launch(void* const* d_in, const int* in_sizes, int n_in,
                              void* d_out, int out_size, void* d_ws, size_t ws_size,
                              hipStream_t stream) {
    (void)in_sizes; (void)n_in; (void)out_size; (void)ws_size;
    const float* x      = (const float*)d_in[0];   // [8,1024,768]
    const float* w_qkv  = (const float*)d_in[1];   // [768,2304]
    const float* w_out  = (const float*)d_in[2];   // [768,768]
    const float* b_out  = (const float*)d_in[3];   // [768]
    float* out = (float*)d_out;                    // [8,1024,768]

    _Float16* x16   = (_Float16*)d_ws;                         // 8192*768
    _Float16* wqkvT = x16   + (size_t)8192 * 768;              // 2304*768
    _Float16* woutT = wqkvT + (size_t)2304 * 768;              // 768*768
    _Float16* q16   = woutT + (size_t)768 * 768;               // 96*1024*64
    _Float16* k16   = q16   + (size_t)BH * NSEQ * DHEAD;
    _Float16* vT16  = k16   + (size_t)BH * NSEQ * DHEAD;
    _Float16* a16   = vT16  + (size_t)BH * NSEQ * DHEAD;       // 8192*768

    const int nX = 8192 * 768;
    f32_to_f16_kernel<<<(nX + 255) / 256, 256, 0, stream>>>(x, x16, nX);
    transpose_f32_to_f16_kernel<<<(768 * 2304 + 255) / 256, 256, 0, stream>>>(
        w_qkv, wqkvT, 768, 2304);
    transpose_f32_to_f16_kernel<<<(768 * 768 + 255) / 256, 256, 0, stream>>>(
        w_out, woutT, 768, 768);

    qkv_gemm_kernel<<<64 * 18, 256, 0, stream>>>(x16, wqkvT, q16, k16, vT16);
    attn_kernel<<<BH * 8, 256, 0, stream>>>(q16, k16, vT16, a16);
    out_gemm_kernel<<<64 * 6, 256, 0, stream>>>(a16, woutT, b_out, out);
}
